// VQVAE_42339787604801
// MI455X (gfx1250) — matrix operand order repaired and checked by reading, not compiled, not run on hardware
//
#include <hip/hip_runtime.h>
#include <hip/hip_fp16.h>
#include <cstdint>
#include <cstddef>

// ---------------------------------------------------------------------------
// Types for CDNA5 WMMA (wave32): A/B = v16h (8 VGPRs), C/D = v8f (8 VGPRs)
// ---------------------------------------------------------------------------
typedef __attribute__((ext_vector_type(16))) _Float16 v16h;
typedef __attribute__((ext_vector_type(8)))  float    v8f;

union F16Frag {
    v16h  v;
    uint4 q[2];
};

__device__ __forceinline__ float lrelu_f(float x) { return x > 0.0f ? x : 0.2f * x; }

// ---------------------------------------------------------------------------
// Workspace layout (bytes). All sizes multiples of 256.
// ---------------------------------------------------------------------------
static constexpr size_t OFF_A     = 0;                        // 16x32x128x128 f32 (33.5MB)  conv1 raw / convT2 raw
static constexpr size_t OFF_B     = OFF_A    + 33554432ull;   // 16x128x128x32 f16 (16.8MB)  NHWC act for conv2 / convT3
static constexpr size_t OFF_C     = OFF_B    + 16777216ull;   // 16x64x64x64 f32 (16.8MB)    raw conv2/conv3/d1 out
static constexpr size_t OFF_D     = OFF_C    + 16777216ull;   // 16x64x64x64 f16 (8.4MB)     NHWC act (t2 / z / zq / d1)
static constexpr size_t OFF_E     = OFF_D    + 8388608ull;    // 16x64x64x64 f32 (16.8MB)    z fp32 NHWC (for loss)
static constexpr size_t OFF_IDX   = OFF_E    + 16777216ull;   // 65536 i32
static constexpr size_t OFF_WF2   = OFF_IDX  + 262144ull;     // conv2 weight frags (64KB)
static constexpr size_t OFF_WF3   = OFF_WF2  + 65536ull;      // conv3 weight frags (72KB)
static constexpr size_t OFF_WFD1  = OFF_WF3  + 73728ull;      // dec conv1 weight frags
static constexpr size_t OFF_CBF   = OFF_WFD1 + 73728ull;      // codebook A-frags (64KB)
static constexpr size_t OFF_CNORM = OFF_CBF  + 65536ull;      // 512 f32 code norms
static constexpr size_t OFF_WFT2  = OFF_CNORM+ 2048ull;       // convT2 parity weight frags (64KB)
static constexpr size_t OFF_WFT3  = OFF_WFT2 + 65536ull;      // convT3 parity weight frags (16KB)
static constexpr size_t OFF_STATS = OFF_WFT3 + 16384ull;      // 5 layers x (sum[128],sq[128]) f32
static constexpr size_t OFF_LOSS  = OFF_STATS+ 5120ull;       // loss accumulator

// ---------------------------------------------------------------------------
// conv1: 3->32, k4 s2 p1, fp32 NCHW in (16,3,256,256) -> raw NCHW (16,32,128,128)
// Tiny K (48), 3 channels -> direct VALU conv (WMMA repack would cost more
// bandwidth than its ~3% FLOP share).
// ---------------------------------------------------------------------------
__global__ __launch_bounds__(256)
void conv1_kernel(const float* __restrict__ x, const float* __restrict__ w,
                  const float* __restrict__ b, float* __restrict__ out)
{
    int i  = blockIdx.x * 256 + threadIdx.x;      // 8388608 total, exact grid
    int wo = i & 127;
    int ho = (i >> 7) & 127;
    int o  = (i >> 14) & 31;
    int n  = i >> 19;
    float acc = b[o];
    #pragma unroll
    for (int kh = 0; kh < 4; ++kh) {
        int hi = ho * 2 - 1 + kh;
        if (hi < 0 || hi >= 256) continue;
        #pragma unroll
        for (int kw = 0; kw < 4; ++kw) {
            int wi = wo * 2 - 1 + kw;
            if (wi < 0 || wi >= 256) continue;
            #pragma unroll
            for (int ci = 0; ci < 3; ++ci) {
                acc += x[(((size_t)n * 3 + ci) * 256 + hi) * 256 + wi] *
                       w[((o * 3 + ci) * 4 + kh) * 4 + kw];
            }
        }
    }
    out[i] = acc;
}

// ---------------------------------------------------------------------------
// BatchNorm statistics: per-(n,c) plane partial sums, atomically merged per c.
// ---------------------------------------------------------------------------
__global__ __launch_bounds__(256)
void bn_stats_kernel(const float* __restrict__ x, float* __restrict__ sums,
                     float* __restrict__ sqs, int C, int HW)
{
    __shared__ float s1[256], s2[256];
    int plane = blockIdx.x;                 // n*C + c
    int c = plane % C;
    const float* p = x + (size_t)plane * HW;
    float a = 0.0f, b = 0.0f;
    for (int i = threadIdx.x; i < HW; i += 256) { float v = p[i]; a += v; b += v * v; }
    s1[threadIdx.x] = a; s2[threadIdx.x] = b;
    __syncthreads();
    for (int s = 128; s > 0; s >>= 1) {
        if ((int)threadIdx.x < s) {
            s1[threadIdx.x] += s1[threadIdx.x + s];
            s2[threadIdx.x] += s2[threadIdx.x + s];
        }
        __syncthreads();
    }
    if (threadIdx.x == 0) { atomicAdd(&sums[c], s1[0]); atomicAdd(&sqs[c], s2[0]); }
}

// ---------------------------------------------------------------------------
// BN normalize + LeakyReLU, NCHW f32 -> NHWC f16 (and optional NHWC f32 copy)
// ---------------------------------------------------------------------------
__global__ __launch_bounds__(256)
void bn_norm_kernel(const float* __restrict__ raw, _Float16* __restrict__ out16,
                    float* __restrict__ out32,
                    const float* __restrict__ sums, const float* __restrict__ sqs,
                    const float* __restrict__ gamma, const float* __restrict__ beta,
                    int C, int H, int W, float invcnt)
{
    size_t i = (size_t)blockIdx.x * 256 + threadIdx.x;     // exact grid
    int w = (int)(i % W); size_t t = i / W;
    int h = (int)(t % H); t /= H;
    int c = (int)(t % C);
    int n = (int)(t / C);
    float mean = sums[c] * invcnt;
    float var  = sqs[c] * invcnt - mean * mean;
    float y = (raw[i] - mean) * rsqrtf(var + 1e-5f);
    y = y * gamma[c] + beta[c];
    y = lrelu_f(y);
    size_t o = (((size_t)n * H + h) * W + w) * (size_t)C + c;
    out16[o] = (_Float16)y;
    if (out32) out32[o] = y;
}

// ---------------------------------------------------------------------------
// Pre-swizzle conv weights (OIHW f32) into WMMA B-fragments:
//   layout [tap][kc][ntile][lane][16 halfs]
//   B (32x16 f16): lane -> col N=lane&15; element e -> K = e + 16*(lane>=16)
// ---------------------------------------------------------------------------
template<int CIN, int COUT, int KH, int KW>
__global__ void prep_wfrag_kernel(const float* __restrict__ w, _Float16* __restrict__ frag)
{
    constexpr int KC = CIN / 32, NT = COUT / 16, TAPS = KH * KW;
    int gid = blockIdx.x * 256 + threadIdx.x;
    if (gid >= TAPS * KC * NT * 32) return;
    int lane = gid & 31;
    int r = gid >> 5;
    int nt = r % NT; r /= NT;
    int kc = r % KC; r /= KC;
    int kh = r / KW, kw = r % KW;
    int o   = nt * 16 + (lane & 15);
    int kbB = (lane & 16) ? 16 : 0;
    _Float16* dst = frag + (size_t)gid * 16;
    #pragma unroll
    for (int e = 0; e < 16; ++e) {
        int ci = kc * 32 + kbB + e;
        dst[e] = (_Float16)w[((o * CIN + ci) * KH + kh) * KW + kw];
    }
}

// ---------------------------------------------------------------------------
// Pre-swizzle ConvTranspose2d weights (CIN,COUT,4,4 f32) into per-parity
// B-fragments. Parity p = (ph<<1)|pw of the OUTPUT pixel fixes which 2x2 tap
// subset applies: kh = ((ph+1)&1) + 2*dh, kw = ((pw+1)&1) + 2*dw.
//   layout [p=4][tap=4][kc][ntile][lane][16 halfs], o >= COUT zero-padded.
// ---------------------------------------------------------------------------
template<int CIN, int COUT, int NT>
__global__ void prep_wfragT_kernel(const float* __restrict__ w, _Float16* __restrict__ frag)
{
    constexpr int KC = CIN / 32;
    int gid = blockIdx.x * 256 + threadIdx.x;
    if (gid >= 4 * 4 * KC * NT * 32) return;
    int lane = gid & 31;
    int r = gid >> 5;
    int nt  = r % NT; r /= NT;
    int kc  = r % KC; r /= KC;
    int tap = r % 4;  r /= 4;
    int p   = r;
    int ph = p >> 1, pw = p & 1;
    int dh = tap >> 1, dw = tap & 1;
    int kh = ((ph + 1) & 1) + 2 * dh;
    int kw = ((pw + 1) & 1) + 2 * dw;
    int o   = nt * 16 + (lane & 15);
    int kbB = (lane & 16) ? 16 : 0;
    _Float16* dst = frag + (size_t)gid * 16;
    #pragma unroll
    for (int e = 0; e < 16; ++e) {
        int ci = kc * 32 + kbB + e;
        float v = (o < COUT) ? w[((ci * COUT + o) * 4 + kh) * 4 + kw] : 0.0f;
        dst[e] = (_Float16)v;
    }
}

// ---------------------------------------------------------------------------
// Pre-swizzle codebook (512,64 f32) into WMMA A-fragments (codes are M rows):
//   layout [tile=32][kc=2][lane][16 halfs]
//   A (16x32 f16): lane -> row M=lane&15; kb=8*(lane>=16);
//                  elements 0..7 -> K=kb..kb+7, 8..15 -> K=kb+16..kb+23
// ---------------------------------------------------------------------------
__global__ void prep_cbfrag_kernel(const float* __restrict__ cb, _Float16* __restrict__ frag)
{
    int gid = blockIdx.x * 256 + threadIdx.x;
    if (gid >= 32 * 2 * 32) return;
    int lane = gid & 31;
    int kc   = (gid >> 5) & 1;
    int tile = gid >> 6;
    int code = tile * 16 + (lane & 15);
    int kb   = (lane & 16) ? 8 : 0;
    _Float16* dst = frag + (size_t)gid * 16;
    #pragma unroll
    for (int e = 0; e < 16; ++e) {
        int k = kc * 32 + kb + ((e < 8) ? e : (e + 8));
        dst[e] = (_Float16)cb[code * 64 + k];
    }
}

__global__ void cnorm_kernel(const float* __restrict__ cb, float* __restrict__ cn)
{
    int k = blockIdx.x * 256 + threadIdx.x;
    if (k >= 512) return;
    float s = 0.0f;
    #pragma unroll
    for (int d = 0; d < 64; ++d) { float v = cb[k * 64 + d]; s += v * v; }
    cn[k] = s;
}

// ---------------------------------------------------------------------------
// Implicit-GEMM conv via v_wmma_f32_16x16x32_f16.
// One wave = 16 consecutive output pixels (M tile) x all COUT channels.
// Activations: NHWC f16. Weights: pre-swizzled fragments. Out: raw NCHW f32 (+bias).
// ---------------------------------------------------------------------------
template<int CIN, int COUT, int KH, int KW, int STRIDE, int PAD,
         int HIN, int WIN, int HOUT, int WOUT>
__global__ __launch_bounds__(256)
void conv_wmma_kernel(const _Float16* __restrict__ act, const uint4* __restrict__ wfrag,
                      const float* __restrict__ bias, float* __restrict__ out)
{
    constexpr int KC = CIN / 32, NT = COUT / 16;
    int wave  = (blockIdx.x * 256 + (int)threadIdx.x) >> 5;
    int lane  = threadIdx.x & 31;
    int mbase = wave * 16;                      // WOUT % 16 == 0 -> tile stays in one row
    int n  = mbase / (HOUT * WOUT);
    int rm = mbase % (HOUT * WOUT);
    int ho = rm / WOUT;
    int wb = rm % WOUT;
    int mrow = lane & 15;
    int wo   = wb + mrow;
    int kbA  = (lane & 16) ? 16 : 0;

    v8f acc[NT];
    #pragma unroll
    for (int t = 0; t < NT; ++t)
        #pragma unroll
        for (int r2 = 0; r2 < 8; ++r2) acc[t][r2] = 0.0f;

    #pragma unroll
    for (int kh = 0; kh < KH; ++kh) {
        int hi = ho * STRIDE - PAD + kh;        // wave-uniform
        if (hi < 0 || hi >= HIN) continue;
        #pragma unroll
        for (int kw = 0; kw < KW; ++kw) {
            int wi = wo * STRIDE - PAD + kw;    // per-lane
            bool ok = (wi >= 0) && (wi < WIN);
            #pragma unroll
            for (int kc = 0; kc < KC; ++kc) {
                F16Frag a;
                if (ok) {
                    const _Float16* s = act + (((size_t)n * HIN + hi) * WIN + wi) * CIN
                                            + kc * 32 + kbA;
                    a.q[0] = *(const uint4*)(s);
                    a.q[1] = *(const uint4*)(s + 8);
                } else {
                    a.q[0] = make_uint4(0u, 0u, 0u, 0u);
                    a.q[1] = make_uint4(0u, 0u, 0u, 0u);
                }
                const uint4* bp = wfrag + (size_t)((((kh * KW + kw) * KC + kc) * NT) * 32 + lane) * 2;
                #pragma unroll
                for (int t = 0; t < NT; ++t) {
                    F16Frag b;
                    b.q[0] = bp[0];
                    b.q[1] = bp[1];
                    bp += 64;                    // next ntile: 32 lanes * 2 uint4
                    acc[t] = __builtin_amdgcn_wmma_f32_16x16x32_f16(
                                 false, a.v, false, b.v, (short)0, acc[t], false, false);
                }
            }
        }
    }

    // C layout: slot r -> (M = r + 8*(lane>=16), N = lane&15). Store NCHW + bias.
    #pragma unroll
    for (int t = 0; t < NT; ++t) {
        int c = t * 16 + (lane & 15);
        float bv = bias[c];
        #pragma unroll
        for (int r2 = 0; r2 < 8; ++r2) {
            int M   = r2 + ((lane & 16) ? 8 : 0);
            int wo2 = wb + M;
            out[(((size_t)n * COUT + c) * HOUT + ho) * WOUT + wo2] = acc[t][r2] + bv;
        }
    }
}

// ---------------------------------------------------------------------------
// Transposed conv k4 s2 p1 via WMMA, parity-decomposed.
// One wave = 16 consecutive pixels of one output-parity subgrid (HIN x WIN)
// x COUT (padded to NT*16). Activations NHWC f16, weights pre-swizzled
// per (parity, tap). Out NCHW f32 (+bias, optional sigmoid).
// ---------------------------------------------------------------------------
template<int CIN, int COUT, int NT, int HIN, int WIN, bool SIG>
__global__ __launch_bounds__(256)
void convt_wmma_kernel(const _Float16* __restrict__ act, const uint4* __restrict__ wfrag,
                       const float* __restrict__ bias, float* __restrict__ out)
{
    constexpr int KC = CIN / 32;
    constexpr int HOUT = 2 * HIN, WOUT = 2 * WIN;
    constexpr int WPP = (16 * HIN * WIN) / 16;       // waves per parity (N=16 batch)
    int wave = (blockIdx.x * 256 + (int)threadIdx.x) >> 5;
    int lane = threadIdx.x & 31;
    int p = wave / WPP;                              // parity class 0..3
    int m = (wave % WPP) * 16;                       // base row in subgrid
    int ph = p >> 1, pw = p & 1;
    int n  = m / (HIN * WIN);
    int rm = m % (HIN * WIN);
    int hb = rm / WIN;
    int wq = rm % WIN;                               // subgrid col base (WIN % 16 == 0)
    int mrow = lane & 15;
    int ho = 2 * hb + ph;                            // wave-uniform
    int wo = 2 * (wq + mrow) + pw;                   // per-lane
    int kbA = (lane & 16) ? 16 : 0;

    v8f acc[NT];
    #pragma unroll
    for (int t = 0; t < NT; ++t)
        #pragma unroll
        for (int r2 = 0; r2 < 8; ++r2) acc[t][r2] = 0.0f;

    #pragma unroll
    for (int dh = 0; dh < 2; ++dh) {
        int kh = ((ph + 1) & 1) + 2 * dh;
        int hi = (ho + 1 - kh) >> 1;                 // wave-uniform
        if (hi < 0 || hi >= HIN) continue;
        #pragma unroll
        for (int dw = 0; dw < 2; ++dw) {
            int kw = ((pw + 1) & 1) + 2 * dw;
            int wi = (wo + 1 - kw) >> 1;             // per-lane
            bool ok = (wi >= 0) && (wi < WIN);
            int tap = dh * 2 + dw;
            #pragma unroll
            for (int kc = 0; kc < KC; ++kc) {
                F16Frag a;
                if (ok) {
                    const _Float16* s = act + (((size_t)n * HIN + hi) * WIN + wi) * CIN
                                            + kc * 32 + kbA;
                    a.q[0] = *(const uint4*)(s);
                    a.q[1] = *(const uint4*)(s + 8);
                } else {
                    a.q[0] = make_uint4(0u, 0u, 0u, 0u);
                    a.q[1] = make_uint4(0u, 0u, 0u, 0u);
                }
                const uint4* bp = wfrag + (size_t)((((p * 4 + tap) * KC + kc) * NT) * 32 + lane) * 2;
                #pragma unroll
                for (int t = 0; t < NT; ++t) {
                    F16Frag b;
                    b.q[0] = bp[0];
                    b.q[1] = bp[1];
                    bp += 64;
                    acc[t] = __builtin_amdgcn_wmma_f32_16x16x32_f16(
                                 false, a.v, false, b.v, (short)0, acc[t], false, false);
                }
            }
        }
    }

    #pragma unroll
    for (int t = 0; t < NT; ++t) {
        int c = t * 16 + (lane & 15);
        if (c < COUT) {
            float bv = bias[c];
            #pragma unroll
            for (int r2 = 0; r2 < 8; ++r2) {
                int M   = r2 + ((lane & 16) ? 8 : 0);
                int wo2 = 2 * (wq + M) + pw;
                float v = acc[t][r2] + bv;
                if (SIG) v = 1.0f / (1.0f + __expf(-v));
                out[(((size_t)n * COUT + c) * HOUT + ho) * WOUT + wo2] = v;
            }
        }
    }
}

// ---------------------------------------------------------------------------
// VQ argmin: scores[z_row, code] = |code|^2 - 2 * z.code  via WMMA
// A = codebook tile (16 codes x K64), B = 16 z rows. One wave = 16 z rows.
// ---------------------------------------------------------------------------
__global__ __launch_bounds__(256)
void vq_argmin_kernel(const _Float16* __restrict__ zf16, const uint4* __restrict__ cbfrag,
                      const float* __restrict__ cnorm, int* __restrict__ idx)
{
    int wave = (blockIdx.x * 256 + (int)threadIdx.x) >> 5;
    int lane = threadIdx.x & 31;
    int rowbase = wave * 16;
    int col  = lane & 15;                        // z row within tile (N)
    int kbB  = (lane & 16) ? 8 : 0;

    const _Float16* zr = zf16 + (size_t)(rowbase + col) * 64 + kbB;
    F16Frag b0, b1;
    b0.q[0] = *(const uint4*)(zr);
    b0.q[1] = *(const uint4*)(zr + 16);
    b1.q[0] = *(const uint4*)(zr + 32);
    b1.q[1] = *(const uint4*)(zr + 48);

    float best = 3.4e38f;
    int   bi   = 0;
    int   koff = (lane & 16) ? 8 : 0;

    for (int t = 0; t < 32; ++t) {
        F16Frag a0, a1;
        const uint4* ap = cbfrag + (size_t)((t * 2 + 0) * 32 + lane) * 2;
        a0.q[0] = ap[0]; a0.q[1] = ap[1];
        ap = cbfrag + (size_t)((t * 2 + 1) * 32 + lane) * 2;
        a1.q[0] = ap[0]; a1.q[1] = ap[1];
        v8f c;
        #pragma unroll
        for (int r = 0; r < 8; ++r) c[r] = 0.0f;
        c = __builtin_amdgcn_wmma_f32_16x16x32_f16(false, a0.v, false, b0.v, (short)0, c, false, false);
        c = __builtin_amdgcn_wmma_f32_16x16x32_f16(false, a1.v, false, b1.v, (short)0, c, false, false);
        #pragma unroll
        for (int r = 0; r < 8; ++r) {
            int code = t * 16 + koff + r;
            float s  = cnorm[code] - 2.0f * c[r];
            if (s < best) { best = s; bi = code; }
        }
    }
    // merge the two half-wave code sets for each z row
    float ob = __shfl_xor(best, 16, 32);
    int   oi = __shfl_xor(bi, 16, 32);
    if (ob < best || (ob == best && oi < bi)) { best = ob; bi = oi; }
    if (lane < 16) idx[rowbase + col] = bi;
}

// ---------------------------------------------------------------------------
// Gather quantized codes (fp32), emit NHWC f16 for decoder, accumulate loss.
// ---------------------------------------------------------------------------
__global__ __launch_bounds__(256)
void vq_gather_kernel(const int* __restrict__ idx, const float* __restrict__ cb,
                      const float* __restrict__ z32, _Float16* __restrict__ zq16,
                      float* __restrict__ loss_acc)
{
    __shared__ float sm[256];
    int i = blockIdx.x * 256 + threadIdx.x;     // 65536*64, exact grid
    int p = i >> 6, c = i & 63;
    float q = cb[idx[p] * 64 + c];
    zq16[i] = (_Float16)q;
    float d = q - z32[i];
    sm[threadIdx.x] = d * d;
    __syncthreads();
    for (int s = 128; s > 0; s >>= 1) {
        if ((int)threadIdx.x < s) sm[threadIdx.x] += sm[threadIdx.x + s];
        __syncthreads();
    }
    if (threadIdx.x == 0) atomicAdd(loss_acc, sm[0]);
}

__global__ void write_idx_kernel(const int* __restrict__ idx, float* __restrict__ out)
{
    int p = blockIdx.x * 256 + threadIdx.x;     // 65536, exact
    out[p] = (float)idx[p];
}

__global__ void finalize_loss_kernel(const float* __restrict__ acc, float* __restrict__ out)
{
    out[0] = 0.25f * acc[0] * (1.0f / (65536.0f * 64.0f));
}

// ---------------------------------------------------------------------------
// Host launcher
// ---------------------------------------------------------------------------
extern "C" void kernel_launch(void* const* d_in, const int* in_sizes, int n_in,
                              void* d_out, int out_size, void* d_ws, size_t ws_size,
                              hipStream_t stream)
{
    (void)in_sizes; (void)n_in; (void)out_size; (void)ws_size;

    const float* x     = (const float*)d_in[0];
    const float* e_w1  = (const float*)d_in[1];
    const float* e_b1  = (const float*)d_in[2];
    const float* e_g1  = (const float*)d_in[3];
    const float* e_be1 = (const float*)d_in[4];
    const float* e_w2  = (const float*)d_in[5];
    const float* e_b2  = (const float*)d_in[6];
    const float* e_g2  = (const float*)d_in[7];
    const float* e_be2 = (const float*)d_in[8];
    const float* e_w3  = (const float*)d_in[9];
    const float* e_b3  = (const float*)d_in[10];
    const float* e_g3  = (const float*)d_in[11];
    const float* e_be3 = (const float*)d_in[12];
    const float* cb    = (const float*)d_in[13];
    const float* d_w1  = (const float*)d_in[14];
    const float* d_b1  = (const float*)d_in[15];
    const float* d_g1  = (const float*)d_in[16];
    const float* d_be1 = (const float*)d_in[17];
    const float* d_w2  = (const float*)d_in[18];
    const float* d_b2  = (const float*)d_in[19];
    const float* d_g2  = (const float*)d_in[20];
    const float* d_be2 = (const float*)d_in[21];
    const float* d_w3  = (const float*)d_in[22];
    const float* d_b3  = (const float*)d_in[23];

    char* ws = (char*)d_ws;
    float*    bufA  = (float*)(ws + OFF_A);
    _Float16* bufB  = (_Float16*)(ws + OFF_B);
    float*    bufC  = (float*)(ws + OFF_C);
    _Float16* bufD  = (_Float16*)(ws + OFF_D);
    float*    bufE  = (float*)(ws + OFF_E);
    int*      idx   = (int*)(ws + OFF_IDX);
    _Float16* wf2   = (_Float16*)(ws + OFF_WF2);
    _Float16* wf3   = (_Float16*)(ws + OFF_WF3);
    _Float16* wfd1  = (_Float16*)(ws + OFF_WFD1);
    _Float16* cbf   = (_Float16*)(ws + OFF_CBF);
    float*    cnrm  = (float*)(ws + OFF_CNORM);
    _Float16* wft2  = (_Float16*)(ws + OFF_WFT2);
    _Float16* wft3  = (_Float16*)(ws + OFF_WFT3);
    float*    stats = (float*)(ws + OFF_STATS);
    float*    lacc  = (float*)(ws + OFF_LOSS);
    float*    out   = (float*)d_out;

    // zero BN stats + loss accumulator (deterministic per call)
    hipMemsetAsync(ws + OFF_STATS, 0, 5 * 256 * 4 + 256, stream);

    // pre-swizzle weights / codebook into WMMA fragment layout
    prep_wfrag_kernel<32, 64, 4, 4><<<8, 256, 0, stream>>>(e_w2, wf2);
    prep_wfrag_kernel<64, 64, 3, 3><<<9, 256, 0, stream>>>(e_w3, wf3);
    prep_wfrag_kernel<64, 64, 3, 3><<<9, 256, 0, stream>>>(d_w1, wfd1);
    prep_wfragT_kernel<64, 32, 2><<<8, 256, 0, stream>>>(d_w2, wft2);   // 4*4*2*2*32 = 2048
    prep_wfragT_kernel<32, 3, 1><<<2, 256, 0, stream>>>(d_w3, wft3);    // 4*4*1*1*32 = 512
    prep_cbfrag_kernel<<<8, 256, 0, stream>>>(cb, cbf);
    cnorm_kernel<<<2, 256, 0, stream>>>(cb, cnrm);

    // ---- encoder stage 1 (direct conv) ----
    conv1_kernel<<<32768, 256, 0, stream>>>(x, e_w1, e_b1, bufA);
    bn_stats_kernel<<<16 * 32, 256, 0, stream>>>(bufA, stats + 0 * 256, stats + 0 * 256 + 128, 32, 128 * 128);
    bn_norm_kernel<<<32768, 256, 0, stream>>>(bufA, bufB, nullptr,
                                              stats + 0 * 256, stats + 0 * 256 + 128,
                                              e_g1, e_be1, 32, 128, 128, 1.0f / (16.0f * 128 * 128));
    // ---- encoder stage 2 (WMMA implicit GEMM) ----
    conv_wmma_kernel<32, 64, 4, 4, 2, 1, 128, 128, 64, 64>
        <<<512, 256, 0, stream>>>(bufB, (const uint4*)wf2, e_b2, bufC);
    bn_stats_kernel<<<16 * 64, 256, 0, stream>>>(bufC, stats + 1 * 256, stats + 1 * 256 + 128, 64, 64 * 64);
    bn_norm_kernel<<<16384, 256, 0, stream>>>(bufC, bufD, nullptr,
                                              stats + 1 * 256, stats + 1 * 256 + 128,
                                              e_g2, e_be2, 64, 64, 64, 1.0f / (16.0f * 64 * 64));
    // ---- encoder stage 3 (WMMA) ----
    conv_wmma_kernel<64, 64, 3, 3, 1, 1, 64, 64, 64, 64>
        <<<512, 256, 0, stream>>>(bufD, (const uint4*)wf3, e_b3, bufC);
    bn_stats_kernel<<<16 * 64, 256, 0, stream>>>(bufC, stats + 2 * 256, stats + 2 * 256 + 128, 64, 64 * 64);
    bn_norm_kernel<<<16384, 256, 0, stream>>>(bufC, bufD, bufE,
                                              stats + 2 * 256, stats + 2 * 256 + 128,
                                              e_g3, e_be3, 64, 64, 64, 1.0f / (16.0f * 64 * 64));
    // ---- vector quantizer (WMMA distance GEMM + argmin) ----
    vq_argmin_kernel<<<512, 256, 0, stream>>>(bufD, (const uint4*)cbf, cnrm, idx);
    vq_gather_kernel<<<16384, 256, 0, stream>>>(idx, cb, bufE, bufD, lacc);
    write_idx_kernel<<<256, 256, 0, stream>>>(idx, out + 3145729);
    finalize_loss_kernel<<<1, 1, 0, stream>>>(lacc, out + 3145728);
    // ---- decoder stage 1 (WMMA) ----
    conv_wmma_kernel<64, 64, 3, 3, 1, 1, 64, 64, 64, 64>
        <<<512, 256, 0, stream>>>(bufD, (const uint4*)wfd1, d_b1, bufC);
    bn_stats_kernel<<<16 * 64, 256, 0, stream>>>(bufC, stats + 3 * 256, stats + 3 * 256 + 128, 64, 64 * 64);
    bn_norm_kernel<<<16384, 256, 0, stream>>>(bufC, bufD, nullptr,
                                              stats + 3 * 256, stats + 3 * 256 + 128,
                                              d_g1, d_be1, 64, 64, 64, 1.0f / (16.0f * 64 * 64));
    // ---- decoder stage 2 (WMMA transposed conv, parity-decomposed) ----
    convt_wmma_kernel<64, 32, 2, 64, 64, false>
        <<<2048, 256, 0, stream>>>(bufD, (const uint4*)wft2, d_b2, bufA);   // 16384 waves
    bn_stats_kernel<<<16 * 32, 256, 0, stream>>>(bufA, stats + 4 * 256, stats + 4 * 256 + 128, 32, 128 * 128);
    bn_norm_kernel<<<32768, 256, 0, stream>>>(bufA, bufB, nullptr,
                                              stats + 4 * 256, stats + 4 * 256 + 128,
                                              d_g2, d_be2, 32, 128, 128, 1.0f / (16.0f * 128 * 128));
    // ---- decoder stage 3 (WMMA transposed conv + sigmoid -> x_recon) ----
    convt_wmma_kernel<32, 3, 1, 128, 128, true>
        <<<8192, 256, 0, stream>>>(bufB, (const uint4*)wft3, d_b3, out);    // 65536 waves
}
